// FAVORAttention_32985348833933
// MI455X (gfx1250) — matrix operand order repaired and checked by reading, not compiled, not run on hardware
//
#include <hip/hip_runtime.h>
#include <hip/hip_bf16.h>
#include <math.h>

// ---------------------------------------------------------------------------
// FAVOR+ (Performer) attention for MI455X (gfx1250), wave32 + WMMA bf16.
// Fragments are fed with 16-byte ds_load_b128 reads (A row-major tiles,
// B tiles staged transposed in LDS so per-lane data is k-contiguous).
// ---------------------------------------------------------------------------

typedef __attribute__((ext_vector_type(16))) __bf16      v16bf;
typedef __attribute__((ext_vector_type(8)))  float       v8f;
typedef __attribute__((ext_vector_type(4)))  unsigned int v4u;

#define B_   4
#define N_   4096
#define D_   1024
#define H_   16
#define DK_  64
#define M_   256
#define BH_  (B_ * H_)

// --------------------------- WMMA helpers ----------------------------------

__device__ __forceinline__ v8f wmma_bf16(v16bf a, v16bf b, v8f c) {
  // D(16x16,f32) = A(16x32,bf16) * B(32x16,bf16) + C
  return __builtin_amdgcn_wmma_f32_16x16x32_bf16(
      /*neg_a=*/false, a, /*neg_b=*/false, b,
      /*c_mod=*/(short)0, c, /*reuse_a=*/false, /*reuse_b=*/false);
}

// A fragment (16x32, M x K) from a row-major LDS tile (leading dim `ld` in
// halves; ld % 8 == 0, k0 % 16 == 0 so both chunks are 16B aligned).
// Lane l: m = l%16, khi = 16*(l/16); halves = [khi..khi+15] (two b128 loads).
__device__ __forceinline__ v16bf frag_a(const __bf16* lds, int row0, int k0, int ld) {
  const int l   = threadIdx.x & 31;
  const int m   = l & 15;
  const int khi = (l >> 4) * 16;
  const __bf16* p = lds + (row0 + m) * ld + k0 + khi;
  union { v4u u[2]; v16bf v; } t;
  t.u[0] = *(const v4u*)(p);
  t.u[1] = *(const v4u*)(p + 8);
  return t.v;
}

// B fragment (32x16, K x N) from a TRANSPOSED LDS tile Bt[n][k] (leading dim
// `ld` in halves; ld % 8 == 0, k0 % 16 == 0). Lane l: n = l%16,
// k = e + 16*(l/16): 16 contiguous halves -> two b128 loads.
__device__ __forceinline__ v16bf frag_bt(const __bf16* lds, int k0, int col0, int ld) {
  const int l  = threadIdx.x & 31;
  const int n  = l & 15;
  const int kb = (l >> 4) << 4;
  const __bf16* p = lds + (col0 + n) * ld + k0 + kb;
  union { v4u u[2]; v16bf v; } t;
  t.u[0] = *(const v4u*)(p);
  t.u[1] = *(const v4u*)(p + 8);
  return t.v;
}

__device__ __forceinline__ v8f v8f_zero() {
  v8f z;
#pragma unroll
  for (int i = 0; i < 8; ++i) z[i] = 0.0f;
  return z;
}

// --------------------------- fp32 -> bf16 convert --------------------------

__global__ __launch_bounds__(256) void cvt_bf16_kernel(
    const float* __restrict__ in, __bf16* __restrict__ out, int n) {
  int i = blockIdx.x * 256 + threadIdx.x;
  if (i < n) out[i] = (__bf16)in[i];
}

// --------------------------- QKV projection GEMM ---------------------------
// C(16384 x 3072) = Xbf(16384 x 1024) * Wqkv^T + bias, scattered into
// Q/K/V in (B,H,N,dk) bf16 layout. B tile staged transposed: Bt[j][k]=W[j,k],
// which is a straight row-copy of W (contiguous both sides).

__global__ __launch_bounds__(256) void qkv_gemm_kernel(
    const __bf16* __restrict__ X,     // (B*N, D)
    const __bf16* __restrict__ W,     // (3D, D) row-major
    const float*  __restrict__ bias,  // (3D)
    __bf16* __restrict__ Qb, __bf16* __restrict__ Kb, __bf16* __restrict__ Vb) {
  constexpr int BM = 128, BN = 128, BK = 32;
  constexpr int LDA = BK + 8, LDB = BK + 8;   // both k-leading, 40 halves
  __shared__ alignas(16) __bf16 As[BM * LDA];
  __shared__ alignas(16) __bf16 Bs[BN * LDB];  // transposed: [n][k]

  const int row0 = blockIdx.x * BM;
  const int col0 = blockIdx.y * BN;
  const int tid  = threadIdx.x;
  const int wid  = tid >> 5;
  const int wr   = wid & 3;    // 4 row-waves * 32 rows
  const int wc   = wid >> 2;   // 2 col-waves * 64 cols

  v8f acc[2][4];
#pragma unroll
  for (int i = 0; i < 2; ++i)
#pragma unroll
    for (int j = 0; j < 4; ++j) acc[i][j] = v8f_zero();

  const int ar = tid >> 1, ak = (tid & 1) * 16;  // 16 contiguous halves/thread
  const int bj = tid >> 1, bk = (tid & 1) * 16;

  for (int k0 = 0; k0 < D_; k0 += BK) {
#pragma unroll
    for (int e = 0; e < 16; ++e)
      As[ar * LDA + ak + e] = X[(size_t)(row0 + ar) * D_ + k0 + ak + e];
#pragma unroll
    for (int e = 0; e < 16; ++e)
      Bs[bj * LDB + bk + e] = W[(size_t)(col0 + bj) * D_ + k0 + bk + e];
    __syncthreads();

    v16bf a0 = frag_a(As, wr * 32 + 0,  0, LDA);
    v16bf a1 = frag_a(As, wr * 32 + 16, 0, LDA);
#pragma unroll
    for (int ct = 0; ct < 4; ++ct) {
      v16bf bfr = frag_bt(Bs, 0, wc * 64 + ct * 16, LDB);
      acc[0][ct] = wmma_bf16(a0, bfr, acc[0][ct]);
      acc[1][ct] = wmma_bf16(a1, bfr, acc[1][ct]);
    }
    __syncthreads();
  }

  const int ln = tid & 15, lh = (tid >> 4) & 1;
#pragma unroll
  for (int rt = 0; rt < 2; ++rt) {
#pragma unroll
    for (int ct = 0; ct < 4; ++ct) {
      int   jg = col0 + wc * 64 + ct * 16 + ln;
      float bv = bias[jg];
      int which = jg >> 10;           // 0:Q 1:K 2:V
      int hd = jg & 1023;
      int h  = hd >> 6, d = hd & 63;
      __bf16* dst = (which == 0) ? Qb : ((which == 1) ? Kb : Vb);
#pragma unroll
      for (int r = 0; r < 8; ++r) {
        int ig = row0 + wr * 32 + rt * 16 + lh * 8 + r;
        int b  = ig >> 12;
        int n  = ig & (N_ - 1);
        dst[((size_t)(b * H_ + h) * N_ + n) * DK_ + d] = (__bf16)(acc[rt][ct][r] + bv);
      }
    }
  }
}

// --------------------------- 0.5 * ||t||^2 per row -------------------------

__global__ __launch_bounds__(256) void sqnorm_kernel(
    const __bf16* __restrict__ T, float* __restrict__ out, int rows) {
  int w    = (blockIdx.x * 256 + threadIdx.x) >> 5;
  int lane = threadIdx.x & 31;
  if (w >= rows) return;
  float a = (float)T[(size_t)w * DK_ + lane];
  float b = (float)T[(size_t)w * DK_ + 32 + lane];
  float s = a * a + b * b;
#pragma unroll
  for (int off = 16; off > 0; off >>= 1) s += __shfl_xor(s, off, 32);
  if (lane == 0) out[w] = 0.5f * s;
}

// --------------------------- phi (feature map) -----------------------------
// For one (b,h), 128 rows: proj = T(128x64) @ omega^T(64x256) via WMMA,
// z = proj - 0.5||t||^2, row-max stabilize, exp, / sqrt(M).
// transp==0: store Qp (BH,N,M). transp==1: store Kp^T (BH,M,N) with PAD mask.
// omega tile staged as Ot[m][k] = straight row-copy of omega (contiguous).

__global__ __launch_bounds__(256) void phi_kernel(
    const __bf16* __restrict__ T,     // (BH, N, 64)
    const __bf16* __restrict__ Om,    // (H, M, 64)
    const float*  __restrict__ sqn,   // (BH, N)
    const unsigned char* __restrict__ mask,  // (B, N), 1 = PAD (K path only)
    __bf16* __restrict__ Out, int transp) {
  constexpr int LDT = DK_ + 8;   // 72
  constexpr int LDO = DK_ + 8;   // transposed omega tile: [m][k]
  __shared__ alignas(16) __bf16 Ts[128 * LDT];
  __shared__ alignas(16) __bf16 Os[M_ * LDO];
  __shared__ float  sq_s[128];
  __shared__ float  vm_s[128];

  const int n0  = blockIdx.x * 128;
  const int bh  = blockIdx.y;
  const int h   = bh & (H_ - 1);
  const int b   = bh >> 4;
  const int tid = threadIdx.x;

  {  // stage T tile (128 x 64): 32 contiguous elems per thread
    int base = tid * 32;
    int r = base >> 6, d0 = base & 63;
    const __bf16* src = T + ((size_t)bh * N_ + n0 + r) * DK_ + d0;
#pragma unroll
    for (int e = 0; e < 32; ++e) Ts[r * LDT + d0 + e] = src[e];
  }
  {  // stage omega (256 x 64) row-copy: thread tid copies row m = tid
    const __bf16* src = Om + ((size_t)h * M_ + tid) * DK_;
#pragma unroll
    for (int k = 0; k < DK_; ++k) Os[tid * LDO + k] = src[k];
  }
  if (tid < 128) {
    sq_s[tid] = sqn[(size_t)bh * N_ + n0 + tid];
    float v = 1.0f;
    if (transp) v = mask[b * N_ + n0 + tid] ? 0.0f : 1.0f;
    vm_s[tid] = v;
  }
  __syncthreads();

  const int wid  = tid >> 5;         // 8 waves * 16 rows
  const int lane = tid & 31;
  const int ln   = lane & 15, lh = lane >> 4;

  v8f acc[16];
#pragma unroll
  for (int i = 0; i < 16; ++i) acc[i] = v8f_zero();

#pragma unroll
  for (int kk = 0; kk < DK_; kk += 32) {
    v16bf a = frag_a(Ts, wid * 16, kk, LDT);
#pragma unroll
    for (int ct = 0; ct < 16; ++ct) {
      v16bf bfr = frag_bt(Os, kk, ct * 16, LDO);
      acc[ct] = wmma_bf16(a, bfr, acc[ct]);
    }
  }

  const float inv_sqrt_m = rsqrtf((float)M_ + 1e-6f);
#pragma unroll
  for (int r = 0; r < 8; ++r) {
    int   rloc = wid * 16 + lh * 8 + r;
    float s    = sq_s[rloc];
    float z[16];
    float mx = -3.0e38f;
#pragma unroll
    for (int ct = 0; ct < 16; ++ct) {
      z[ct] = acc[ct][r] - s;
      mx = fmaxf(mx, z[ct]);
    }
    // reduce row-max across the 16 lanes holding this row's columns
#pragma unroll
    for (int off = 8; off > 0; off >>= 1) mx = fmaxf(mx, __shfl_xor(mx, off, 32));
    float scale = vm_s[rloc] * inv_sqrt_m;
    int   ng    = n0 + rloc;
#pragma unroll
    for (int ct = 0; ct < 16; ++ct) {
      float p = __expf(z[ct] - mx) * scale;
      int   m = ct * 16 + ln;
      if (!transp)
        Out[((size_t)bh * N_ + ng) * M_ + m] = (__bf16)p;
      else
        Out[((size_t)bh * M_ + m) * N_ + ng] = (__bf16)p;
    }
  }
}

// --------------------------- ksum = sum_n Kp -------------------------------

__global__ __launch_bounds__(256) void ksum_kernel(
    const __bf16* __restrict__ KpT, float* __restrict__ ks) {
  int w    = (blockIdx.x * 256 + threadIdx.x) >> 5;
  int lane = threadIdx.x & 31;
  if (w >= BH_ * M_) return;
  const __bf16* p = KpT + (size_t)w * N_;
  float s = 0.0f;
  for (int i = lane; i < N_; i += 32) s += (float)p[i];
#pragma unroll
  for (int off = 16; off > 0; off >>= 1) s += __shfl_xor(s, off, 32);
  if (lane == 0) ks[w] = s;
}

// --------------------------- KV = Kp^T @ V (+ksum pack) --------------------
// Per (b,h): C(256 x 64) contracted over N=4096. Output is the padded
// (256 x 80) matrix [KV | ksum | 0...] used as B by the num GEMM.
// V tile staged transposed: Vt[d][n-chunk].

__global__ __launch_bounds__(256) void kv_gemm_kernel(
    const __bf16* __restrict__ KpT,  // (BH, M, N)
    const __bf16* __restrict__ Vb,   // (BH, N, 64)
    const float*  __restrict__ ks,   // (BH, M)
    __bf16* __restrict__ KVp) {      // (BH, M, 80)
  constexpr int BK = 64;
  constexpr int LDA = BK + 8;        // 72
  constexpr int LDB = BK + 8;        // transposed V tile: [d][k], 72
  __shared__ alignas(16) __bf16 As[M_ * LDA];
  __shared__ alignas(16) __bf16 Bs[DK_ * LDB];

  const int bh  = blockIdx.x;
  const int tid = threadIdx.x;
  const int wid = tid >> 5;          // 8 waves * 32 rows = 256 rows

  v8f acc[2][4];
#pragma unroll
  for (int i = 0; i < 2; ++i)
#pragma unroll
    for (int j = 0; j < 4; ++j) acc[i][j] = v8f_zero();

  for (int n0 = 0; n0 < N_; n0 += BK) {
    {  // A: thread tid loads 64 contiguous from Kp^T row m=tid
      const __bf16* src = KpT + ((size_t)bh * M_ + tid) * N_ + n0;
#pragma unroll
      for (int e = 0; e < BK; ++e) As[tid * LDA + e] = src[e];
    }
    {  // B: 64x64 tile of V, transposed into Bs[d][k]
      int base = tid * 16;
      int k = base >> 6, d0 = base & 63;
      const __bf16* src = Vb + ((size_t)bh * N_ + n0 + k) * DK_ + d0;
#pragma unroll
      for (int e = 0; e < 16; ++e) Bs[(d0 + e) * LDB + k] = src[e];
    }
    __syncthreads();
#pragma unroll
    for (int kk = 0; kk < BK; kk += 32) {
      v16bf a0 = frag_a(As, wid * 32 + 0,  kk, LDA);
      v16bf a1 = frag_a(As, wid * 32 + 16, kk, LDA);
#pragma unroll
      for (int ct = 0; ct < 4; ++ct) {
        v16bf bfr = frag_bt(Bs, kk, ct * 16, LDB);
        acc[0][ct] = wmma_bf16(a0, bfr, acc[0][ct]);
        acc[1][ct] = wmma_bf16(a1, bfr, acc[1][ct]);
      }
    }
    __syncthreads();
  }

  const int ln = tid & 15, lh = (tid >> 4) & 1;
#pragma unroll
  for (int rt = 0; rt < 2; ++rt)
#pragma unroll
    for (int ct = 0; ct < 4; ++ct)
#pragma unroll
      for (int r = 0; r < 8; ++r) {
        int m = wid * 32 + rt * 16 + lh * 8 + r;
        int d = ct * 16 + ln;
        KVp[((size_t)bh * M_ + m) * 80 + d] = (__bf16)acc[rt][ct][r];
      }
  // column 64 = ksum, columns 65..79 = 0 (thread tid handles row m = tid)
  {
    __bf16* row = KVp + ((size_t)bh * M_ + tid) * 80;
    row[64] = (__bf16)ks[(size_t)bh * M_ + tid];
#pragma unroll
    for (int c = 65; c < 80; ++c) row[c] = (__bf16)0.0f;
  }
}

// --------------------------- num/den GEMM + normalize ----------------------
// Per (b,h): C(128 x 80) = Qp(128 x 256) @ KVp(256 x 80).
// Column 64 is the denominator; ctx = num / (den + eps), merged-head bf16.
// KVp tile staged transposed: Bt[j][k].

__global__ __launch_bounds__(256) void num_gemm_kernel(
    const __bf16* __restrict__ Qp,   // (BH, N, M)
    const __bf16* __restrict__ KVp,  // (BH, M, 80)
    __bf16* __restrict__ Ctx) {      // (B, N, D)
  constexpr int BM = 128, BN = 80, BK = 32;
  constexpr int LDA = BK + 8, LDB = BK + 8;   // 40
  __shared__ alignas(16) __bf16 As[BM * LDA];
  __shared__ alignas(16) __bf16 Bs[BN * LDB]; // transposed: [j][k]
  __shared__ float  den_s[BM];

  const int n0  = blockIdx.x * BM;
  const int bh  = blockIdx.y;
  const int h   = bh & (H_ - 1);
  const int b   = bh >> 4;
  const int tid = threadIdx.x;
  const int wid = tid >> 5;          // 8 waves * 16 rows

  v8f acc[5];
#pragma unroll
  for (int i = 0; i < 5; ++i) acc[i] = v8f_zero();

  const int ar = tid >> 1, ak = (tid & 1) * 16;

  for (int k0 = 0; k0 < M_; k0 += BK) {
#pragma unroll
    for (int e = 0; e < 16; ++e)
      As[ar * LDA + ak + e] = Qp[((size_t)bh * N_ + n0 + ar) * M_ + k0 + ak + e];
    for (int idx = tid; idx < BK * BN; idx += 256) {
      int k = idx / BN, j = idx % BN;          // global coalesced over j
      Bs[j * LDB + k] = KVp[((size_t)bh * M_ + k0 + k) * 80 + j];
    }
    __syncthreads();
    v16bf a = frag_a(As, wid * 16, 0, LDA);
#pragma unroll
    for (int ct = 0; ct < 5; ++ct) {
      v16bf bfr = frag_bt(Bs, 0, ct * 16, LDB);
      acc[ct] = wmma_bf16(a, bfr, acc[ct]);
    }
    __syncthreads();
  }

  const int ln = tid & 15, lh = (tid >> 4) & 1;
  if (ln == 0) {  // lane holding column 64 of fragment 4 publishes den per row
#pragma unroll
    for (int r = 0; r < 8; ++r) den_s[wid * 16 + lh * 8 + r] = acc[4][r];
  }
  __syncthreads();

#pragma unroll
  for (int ct = 0; ct < 4; ++ct)
#pragma unroll
    for (int r = 0; r < 8; ++r) {
      int   rloc = wid * 16 + lh * 8 + r;
      float den  = den_s[rloc] + 1e-6f;
      float val  = acc[ct][r] / den;
      int   d    = ct * 16 + ln;
      int   n    = n0 + rloc;
      Ctx[(size_t)(b * N_ + n) * D_ + h * DK_ + d] = (__bf16)val;
    }
}

// --------------------------- output projection GEMM ------------------------

__global__ __launch_bounds__(256) void out_gemm_kernel(
    const __bf16* __restrict__ Ctx,   // (B*N, D)
    const __bf16* __restrict__ W,     // (D, D) row-major
    const float*  __restrict__ bias,  // (D)
    const unsigned char* __restrict__ mask,  // (B,N), 1 = PAD
    float* __restrict__ Out) {        // (B*N, D) fp32
  constexpr int BM = 128, BN = 128, BK = 32;
  constexpr int LDA = BK + 8, LDB = BK + 8;
  __shared__ alignas(16) __bf16 As[BM * LDA];
  __shared__ alignas(16) __bf16 Bs[BN * LDB]; // transposed: [j][k]

  const int row0 = blockIdx.x * BM;
  const int col0 = blockIdx.y * BN;
  const int tid  = threadIdx.x;
  const int wid  = tid >> 5;
  const int wr   = wid & 3;
  const int wc   = wid >> 2;

  v8f acc[2][4];
#pragma unroll
  for (int i = 0; i < 2; ++i)
#pragma unroll
    for (int j = 0; j < 4; ++j) acc[i][j] = v8f_zero();

  const int ar = tid >> 1, ak = (tid & 1) * 16;
  const int bj = tid >> 1, bk = (tid & 1) * 16;

  for (int k0 = 0; k0 < D_; k0 += BK) {
#pragma unroll
    for (int e = 0; e < 16; ++e)
      As[ar * LDA + ak + e] = Ctx[(size_t)(row0 + ar) * D_ + k0 + ak + e];
#pragma unroll
    for (int e = 0; e < 16; ++e)
      Bs[bj * LDB + bk + e] = W[(size_t)(col0 + bj) * D_ + k0 + bk + e];
    __syncthreads();

    v16bf a0 = frag_a(As, wr * 32 + 0,  0, LDA);
    v16bf a1 = frag_a(As, wr * 32 + 16, 0, LDA);
#pragma unroll
    for (int ct = 0; ct < 4; ++ct) {
      v16bf bfr = frag_bt(Bs, 0, wc * 64 + ct * 16, LDB);
      acc[0][ct] = wmma_bf16(a0, bfr, acc[0][ct]);
      acc[1][ct] = wmma_bf16(a1, bfr, acc[1][ct]);
    }
    __syncthreads();
  }

  const int ln = tid & 15, lh = (tid >> 4) & 1;
#pragma unroll
  for (int rt = 0; rt < 2; ++rt)
#pragma unroll
    for (int ct = 0; ct < 4; ++ct) {
      int   jg = col0 + wc * 64 + ct * 16 + ln;
      float bv = bias[jg];
#pragma unroll
      for (int r = 0; r < 8; ++r) {
        int ig = row0 + wr * 32 + rt * 16 + lh * 8 + r;
        int b  = ig >> 12;
        int n  = ig & (N_ - 1);
        float val = mask[b * N_ + n] ? 0.0f : (acc[rt][ct][r] + bv);
        Out[(size_t)ig * D_ + jg] = val;
      }
    }
}

// --------------------------- host launcher ---------------------------------

extern "C" void kernel_launch(void* const* d_in, const int* in_sizes, int n_in,
                              void* d_out, int out_size, void* d_ws, size_t ws_size,
                              hipStream_t stream) {
  (void)in_sizes; (void)n_in; (void)out_size; (void)ws_size;

  const float*         x     = (const float*)d_in[0];
  const unsigned char* mask  = (const unsigned char*)d_in[1];
  const float*         Wqkv  = (const float*)d_in[2];
  const float*         bqkv  = (const float*)d_in[3];
  const float*         Wout  = (const float*)d_in[4];
  const float*         bout  = (const float*)d_in[5];
  const float*         omega = (const float*)d_in[6];
  float*               out   = (float*)d_out;

  char*  base = (char*)d_ws;
  size_t off  = 0;
  auto take = [&](size_t bytes) -> void* {
    void* p = base + off;
    off += (bytes + 255) & ~(size_t)255;
    return p;
  };

  const size_t nX  = (size_t)B_ * N_ * D_;        // 16,777,216
  const size_t nW1 = (size_t)3 * D_ * D_;
  const size_t nW2 = (size_t)D_ * D_;
  const size_t nOm = (size_t)H_ * M_ * DK_;
  const size_t nT  = (size_t)BH_ * N_ * DK_;      // per Q/K/V tensor
  const size_t nP  = (size_t)BH_ * N_ * M_;       // per Qp/KpT tensor
  const size_t nR  = (size_t)BH_ * N_;            // per sqnorm tensor

  __bf16* xbf    = (__bf16*)take(nX  * 2);  // reused as Qp after QKV GEMM
  __bf16* wqkvbf = (__bf16*)take(nW1 * 2);
  __bf16* woutbf = (__bf16*)take(nW2 * 2);
  __bf16* ombf   = (__bf16*)take(nOm * 2);
  __bf16* qbf    = (__bf16*)take(nT  * 2);  // reused as Ctx after phi(Q)
  __bf16* kbf    = (__bf16*)take(nT  * 2);
  __bf16* vbf    = (__bf16*)take(nT  * 2);
  __bf16* KpT    = (__bf16*)take(nP  * 2);
  float*  sqnQ   = (float*)take(nR * 4);
  float*  sqnK   = (float*)take(nR * 4);
  float*  ksum   = (float*)take((size_t)BH_ * M_ * 4);
  __bf16* KVp    = (__bf16*)take((size_t)BH_ * M_ * 80 * 2);

  __bf16* Qp  = xbf;   // alias: x only read by QKV GEMM
  __bf16* Ctx = qbf;   // alias: Q only read by sqnorm + phi(Q)

  // 1) fp32 -> bf16 staging
  cvt_bf16_kernel<<<dim3((unsigned)((nX  + 255) / 256)), 256, 0, stream>>>(x,     xbf,    (int)nX);
  cvt_bf16_kernel<<<dim3((unsigned)((nW1 + 255) / 256)), 256, 0, stream>>>(Wqkv,  wqkvbf, (int)nW1);
  cvt_bf16_kernel<<<dim3((unsigned)((nW2 + 255) / 256)), 256, 0, stream>>>(Wout,  woutbf, (int)nW2);
  cvt_bf16_kernel<<<dim3((unsigned)((nOm + 255) / 256)), 256, 0, stream>>>(omega, ombf,   (int)nOm);

  // 2) QKV projection (WMMA), scatter into (B,H,N,dk)
  qkv_gemm_kernel<<<dim3(128, 24), 256, 0, stream>>>(xbf, wqkvbf, bqkv, qbf, kbf, vbf);

  // 3) 0.5*||t||^2 rows
  {
    int rows = (int)nR;
    int blocks = (rows * 32 + 255) / 256;
    sqnorm_kernel<<<dim3(blocks), 256, 0, stream>>>(qbf, sqnQ, rows);
    sqnorm_kernel<<<dim3(blocks), 256, 0, stream>>>(kbf, sqnK, rows);
  }

  // 4) feature maps (WMMA): Qp row-major, Kp transposed + PAD-masked
  phi_kernel<<<dim3(N_ / 128, BH_), 256, 0, stream>>>(qbf, ombf, sqnQ, mask, Qp, 0);
  phi_kernel<<<dim3(N_ / 128, BH_), 256, 0, stream>>>(kbf, ombf, sqnK, mask, KpT, 1);

  // 5) ksum over n
  {
    int waves = BH_ * M_;
    ksum_kernel<<<dim3((waves * 32 + 255) / 256), 256, 0, stream>>>(KpT, ksum);
  }

  // 6) KV = Kp^T @ V (WMMA) + pack [KV | ksum | 0]
  kv_gemm_kernel<<<dim3(BH_), 256, 0, stream>>>(KpT, vbf, ksum, KVp);

  // 7) num/den GEMM (WMMA) + normalize -> merged-head ctx
  num_gemm_kernel<<<dim3(N_ / 128, BH_), 256, 0, stream>>>(Qp, KVp, Ctx);

  // 8) output projection (WMMA) + bias + PAD zeroing, fp32 out
  out_gemm_kernel<<<dim3(128, 8), 256, 0, stream>>>(Ctx, woutbf, bout, mask, out);
}
